// MambaBlock_68341519614419
// MI455X (gfx1250) — matrix operand order repaired and checked by reading, not compiled, not run on hardware
//
#include <hip/hip_runtime.h>
#include <math.h>

typedef __attribute__((ext_vector_type(16))) __bf16 v16bf;
typedef __attribute__((ext_vector_type(8)))  __bf16 v8bf;
typedef __attribute__((ext_vector_type(8)))  float  v8f;

#define BATCH    2
#define SEQLEN   1024
#define DIN      512
#define DMODEL   1024
#define DSTATE   16
#define DDISCR   64
#define KERW     4
#define NTOK     (BATCH * SEQLEN)       // 2048

// ---------------------------------------------------------------------------
// f32 -> bf16 conversion (grid-stride)
// ---------------------------------------------------------------------------
__global__ void cvt_f32_bf16(const float* __restrict__ in,
                             __bf16* __restrict__ out, int n) {
    int stride = gridDim.x * blockDim.x;
    for (int i = blockIdx.x * blockDim.x + threadIdx.x; i < n; i += stride)
        out[i] = (__bf16)in[i];
}

// ---------------------------------------------------------------------------
// NT GEMM via v_wmma_f32_16x16x32_bf16.
// C[M,N] = A[M,K] * W[N,K]^T.  A, W are bf16 with K contiguous; C is f32.
// One wave -> 16x64 strip of C (4 accumulators share one A fragment).
// 8 waves per block stacked along M -> block tile 128x64.
// Fragments load directly from global (everything is L2-resident: 192MB L2).
// ---------------------------------------------------------------------------
__global__ __launch_bounds__(256)
void gemm_nt_wmma(const __bf16* __restrict__ A,
                  const __bf16* __restrict__ W,
                  float* __restrict__ C,
                  int M, int N, int K) {
    const int lane = threadIdx.x & 31;
    const int wave = threadIdx.x >> 5;
    const int row  = lane & 15;
    const int hi   = lane >> 4;          // 0: lanes 0-15, 1: lanes 16-31

    const int mBase = blockIdx.y * 128 + wave * 16;
    const int nBase = blockIdx.x * 64;
    if (mBase >= M || nBase >= N) return;

    const __bf16* pA  = A + (size_t)(mBase + row) * K;
    const __bf16* pW0 = W + (size_t)(nBase + row +  0) * K;
    const __bf16* pW1 = W + (size_t)(nBase + row + 16) * K;
    const __bf16* pW2 = W + (size_t)(nBase + row + 32) * K;
    const __bf16* pW3 = W + (size_t)(nBase + row + 48) * K;

    v8f acc0 = {}, acc1 = {}, acc2 = {}, acc3 = {};

    for (int k0 = 0; k0 < K; k0 += 32) {
        // A fragment (16x32 MxK), contiguous k order used here (same as B):
        //   lanes 0-15 : row m, K = {k0..k0+15}
        //   lanes16-31 : row m, K = {k0+16..k0+31}
        v16bf af = *(const v16bf*)(pA + k0 + hi * 16);

        // B fragment (32x16 KxN): lane = column n, 16 consecutive K values,
        // lanes 16-31 take the upper K half. W rows are exactly those columns.
        const int kb = k0 + hi * 16;
        v16bf b0 = *(const v16bf*)(pW0 + kb);
        v16bf b1 = *(const v16bf*)(pW1 + kb);
        v16bf b2 = *(const v16bf*)(pW2 + kb);
        v16bf b3 = *(const v16bf*)(pW3 + kb);

        __builtin_prefetch((const void*)(pA + k0 + 128), 0, 1);

        acc0 = __builtin_amdgcn_wmma_f32_16x16x32_bf16(false, af, false, b0,
                                                       (short)0, acc0, false, false);
        acc1 = __builtin_amdgcn_wmma_f32_16x16x32_bf16(false, af, false, b1,
                                                       (short)0, acc1, false, false);
        acc2 = __builtin_amdgcn_wmma_f32_16x16x32_bf16(false, af, false, b2,
                                                       (short)0, acc2, false, false);
        acc3 = __builtin_amdgcn_wmma_f32_16x16x32_bf16(false, af, false, b3,
                                                       (short)0, acc3, false, false);
    }

    // C/D layout: VGPR r -> (M = r + 8*hi, N = lane&15)
    const int cn = nBase + (lane & 15);
    const int rb = mBase + hi * 8;
#pragma unroll
    for (int r = 0; r < 8; ++r) {
        float* crow = C + (size_t)(rb + r) * N;
        crow[cn +  0] = acc0[r];
        crow[cn + 16] = acc1[r];
        crow[cn + 32] = acc2[r];
        crow[cn + 48] = acc3[r];
    }
}

// ---------------------------------------------------------------------------
// Causal depthwise conv (K=4) + SiLU, reading the 'a' half of ab in place.
// ---------------------------------------------------------------------------
__global__ void conv_silu(const float* __restrict__ ab,
                          const float* __restrict__ w,
                          const float* __restrict__ bias,
                          float* __restrict__ x) {
    int idx = blockIdx.x * blockDim.x + threadIdx.x;  // over B*L*DMODEL
    if (idx >= NTOK * DMODEL) return;
    int d = idx % DMODEL;
    int l = (idx / DMODEL) % SEQLEN;
    int b = idx / (DMODEL * SEQLEN);

    const float* col = ab + (size_t)b * SEQLEN * (2 * DMODEL) + d;
    float acc = bias[d];
#pragma unroll
    for (int k = 0; k < KERW; ++k) {
        int ls = l - (KERW - 1) + k;
        if (ls >= 0)
            acc = fmaf(w[d * KERW + k], col[(size_t)ls * (2 * DMODEL)], acc);
    }
    x[idx] = acc / (1.0f + __expf(-acc));   // SiLU
}

// ---------------------------------------------------------------------------
// Per-token projections: Bm, Cm (16 each) and delta = softplus(D + x W_D1^T W_D2^T).
// One block (256 threads) per token; x staged in LDS.
// ---------------------------------------------------------------------------
__global__ __launch_bounds__(256)
void proj_small(const float* __restrict__ x,
                const float* __restrict__ W_B,  const float* __restrict__ W_C,
                const float* __restrict__ W_D1, const float* __restrict__ W_D2,
                const float* __restrict__ Dv,
                float* __restrict__ Bm, float* __restrict__ Cm,
                float* __restrict__ delta) {
    __shared__ float xs[DMODEL];
    __shared__ float ered[4][DDISCR];
    __shared__ float e[DDISCR];
    __shared__ float bcp[2][16][16];

    const int t = threadIdx.x;
    const size_t tok = blockIdx.x;
    const float* xr = x + tok * DMODEL;

    for (int i = t; i < DMODEL; i += 256) xs[i] = xr[i];
    __syncthreads();

    // e = x @ W_D1^T   (64 outputs, split K=1024 into 4 parts)
    {
        int j = t & 63, part = t >> 6;
        const float* wr = W_D1 + (size_t)j * DMODEL + part * 256;
        const float* xv = xs + part * 256;
        float p = 0.f;
        for (int k = 0; k < 256; ++k) p = fmaf(wr[k], xv[k], p);
        ered[part][j] = p;
    }
    // Bm, Cm (16 outputs each, split K into 16 parts)
    {
        int s = t & 15, part = t >> 4;
        const float* wb = W_B + (size_t)s * DMODEL + part * 64;
        const float* wc = W_C + (size_t)s * DMODEL + part * 64;
        const float* xv = xs + part * 64;
        float pb = 0.f, pc = 0.f;
        for (int k = 0; k < 64; ++k) {
            pb = fmaf(wb[k], xv[k], pb);
            pc = fmaf(wc[k], xv[k], pc);
        }
        __syncthreads();
        bcp[0][part][s] = pb;
        bcp[1][part][s] = pc;
    }
    __syncthreads();
    if (t < 64) e[t] = ered[0][t] + ered[1][t] + ered[2][t] + ered[3][t];
    if (t < 16) {
        float sb = 0.f;
        for (int p = 0; p < 16; ++p) sb += bcp[0][p][t];
        Bm[tok * DSTATE + t] = sb;
    } else if (t < 32) {
        int s = t - 16;
        float sc = 0.f;
        for (int p = 0; p < 16; ++p) sc += bcp[1][p][s];
        Cm[tok * DSTATE + s] = sc;
    }
    __syncthreads();

    // delta[d] = softplus(D[d] + e . W_D2[d,:])
#pragma unroll
    for (int i = 0; i < 4; ++i) {
        int d = t + i * 256;
        const float* w2 = W_D2 + (size_t)d * DDISCR;
        float acc = Dv[d];
        for (int j = 0; j < DDISCR; ++j) acc = fmaf(w2[j], e[j], acc);
        delta[tok * DMODEL + d] = (acc > 20.f) ? acc : log1pf(__expf(acc));
    }
}

// ---------------------------------------------------------------------------
// Sequential SSM scan fused with output projection epilogue:
//   h[l] = (exp(-A)*delta)*h[l-1] + Bm*delta*x ;  y = <h,Cm> + D*x
//   gated = (y * silu(b_gate))  -> bf16 for the final WMMA GEMM.
// One wave = 2 channels x 16 states (wave32). 1024 waves total.
// 16-lane butterfly shuffle reduces over the state dim.
// ---------------------------------------------------------------------------
__global__ __launch_bounds__(256)
void scan_gate(const float* __restrict__ x,  const float* __restrict__ delta,
               const float* __restrict__ Bm, const float* __restrict__ Cm,
               const float* __restrict__ A,  const float* __restrict__ Dv,
               const float* __restrict__ ab, __bf16* __restrict__ gated) {
    const int w    = (blockIdx.x * blockDim.x + threadIdx.x) >> 5;  // wave id
    const int lane = threadIdx.x & 31;
    const int s    = lane & 15;
    const int g    = lane >> 4;
    const int b    = w >> 9;                       // 512 wave-pairs per batch
    const int d    = ((w & 511) << 1) | g;

    const float ena = __expf(-A[d * DSTATE + s]);
    const float dd  = Dv[d];
    float h = 0.f;

    for (int l = 0; l < SEQLEN; ++l) {
        const size_t tok = (size_t)b * SEQLEN + l;
        float xv = x[tok * DMODEL + d];
        float dl = delta[tok * DMODEL + d];
        float bm = Bm[tok * DSTATE + s];
        float cm = Cm[tok * DSTATE + s];

        h = fmaf(ena * dl, h, bm * dl * xv);
        float ys = h * cm;
        ys += __shfl_xor(ys, 1, 32);
        ys += __shfl_xor(ys, 2, 32);
        ys += __shfl_xor(ys, 4, 32);
        ys += __shfl_xor(ys, 8, 32);

        if (s == 0) {
            float y  = ys + dd * xv;
            float bg = ab[tok * (2 * DMODEL) + DMODEL + d];
            float gt = bg / (1.0f + __expf(-bg));
            gated[tok * DMODEL + d] = (__bf16)(y * gt);
        }
    }
}

// ---------------------------------------------------------------------------
extern "C" void kernel_launch(void* const* d_in, const int* in_sizes, int n_in,
                              void* d_out, int out_size, void* d_ws, size_t ws_size,
                              hipStream_t stream) {
    (void)in_sizes; (void)n_in; (void)out_size; (void)ws_size;

    const float* seq   = (const float*)d_in[0];
    const float* W_in  = (const float*)d_in[1];
    const float* W_out = (const float*)d_in[2];
    const float* W_B   = (const float*)d_in[3];
    const float* W_C   = (const float*)d_in[4];
    const float* W_D1  = (const float*)d_in[5];
    const float* W_D2  = (const float*)d_in[6];
    const float* convw = (const float*)d_in[7];
    const float* convb = (const float*)d_in[8];
    const float* Amat  = (const float*)d_in[9];
    const float* Dvec  = (const float*)d_in[10];

    char* wsp = (char*)d_ws;
    auto alloc = [&](size_t bytes) -> char* {
        char* p = wsp;
        wsp += (bytes + 255) & ~(size_t)255;
        return p;
    };
    __bf16* seq_bf  = (__bf16*)alloc((size_t)NTOK * DIN * 2);          // 2 MB
    __bf16* Win_bf  = (__bf16*)alloc((size_t)2 * DMODEL * DIN * 2);    // 2 MB
    __bf16* Wout_bf = (__bf16*)alloc((size_t)DIN * DMODEL * 2);        // 1 MB
    float*  ab      = (float*) alloc((size_t)NTOK * 2 * DMODEL * 4);   // 16 MB
    float*  x       = (float*) alloc((size_t)NTOK * DMODEL * 4);       // 8 MB
    float*  delta   = (float*) alloc((size_t)NTOK * DMODEL * 4);       // 8 MB
    float*  Bm      = (float*) alloc((size_t)NTOK * DSTATE * 4);
    float*  Cm      = (float*) alloc((size_t)NTOK * DSTATE * 4);
    __bf16* gated   = (__bf16*)alloc((size_t)NTOK * DMODEL * 2);       // 4 MB

    // bf16 conversions for the WMMA GEMM operands
    cvt_f32_bf16<<<512, 256, 0, stream>>>(seq,   seq_bf,  NTOK * DIN);
    cvt_f32_bf16<<<512, 256, 0, stream>>>(W_in,  Win_bf,  2 * DMODEL * DIN);
    cvt_f32_bf16<<<512, 256, 0, stream>>>(W_out, Wout_bf, DIN * DMODEL);

    // GEMM1: ab[2048,2048] = seq[2048,512] @ W_in[2048,512]^T
    gemm_nt_wmma<<<dim3((2 * DMODEL) / 64, NTOK / 128), 256, 0, stream>>>(
        seq_bf, Win_bf, ab, NTOK, 2 * DMODEL, DIN);

    // x = silu(causal depthwise conv(a-half of ab))
    conv_silu<<<(NTOK * DMODEL) / 256, 256, 0, stream>>>(ab, convw, convb, x);

    // Bm, Cm, delta
    proj_small<<<NTOK, 256, 0, stream>>>(x, W_B, W_C, W_D1, W_D2, Dvec,
                                         Bm, Cm, delta);

    // scan + state reduction + gate + bf16 cvt (1024 waves)
    scan_gate<<<128, 256, 0, stream>>>(x, delta, Bm, Cm, Amat, Dvec, ab, gated);

    // GEMM2: out[2048,512] = gated[2048,1024] @ W_out[512,1024]^T
    gemm_nt_wmma<<<dim3(DIN / 64, NTOK / 128), 256, 0, stream>>>(
        gated, Wout_bf, (float*)d_out, NTOK, DIN, DMODEL);
}
